// Seq2SeqAttention_9088150798964
// MI455X (gfx1250) — matrix operand order repaired and checked by reading, not compiled, not run on hardware
//
#include <hip/hip_runtime.h>
#include <hip/hip_bf16.h>
#include <math.h>

typedef __bf16 bf16_t;
typedef __attribute__((ext_vector_type(16))) __bf16 v16bf;
typedef __attribute__((ext_vector_type(8)))  __bf16 v8bf;
typedef __attribute__((ext_vector_type(8)))  float  v8f;

#define S_LEN 50
#define BATCH 64
#define EMB   256
#define HENC  512
#define VOUT  32000
#define KFC   1792          /* 512 + 1024 + 256 */

#if __has_builtin(__builtin_amdgcn_global_load_async_to_lds_b128)
#define HAVE_ASYNC_LDS 1
typedef int v4i_ __attribute__((vector_size(16)));
typedef __attribute__((address_space(1))) v4i_* gas1_t;
typedef __attribute__((address_space(3))) v4i_* las3_t;
#else
#define HAVE_ASYNC_LDS 0
#endif

static __device__ __forceinline__ void wait_async0() {
#if __has_builtin(__builtin_amdgcn_s_wait_asynccnt)
  __builtin_amdgcn_s_wait_asynccnt(0);
#else
  asm volatile("s_wait_asynccnt 0" ::: "memory");
#endif
}

static __device__ __forceinline__ float sigmoidf_(float x) {
  return 1.0f / (1.0f + __expf(-x));
}

/* ---------------- elementwise helpers ---------------- */

__global__ void k_cvt_bf16(const float* __restrict__ s, bf16_t* __restrict__ d, long n) {
  long i = (long)blockIdx.x * blockDim.x + threadIdx.x;
  long stride = (long)gridDim.x * blockDim.x;
  for (; i < n; i += stride) d[i] = (bf16_t)s[i];
}

__global__ void k_zero_f32(float* p, long n) {
  long i = (long)blockIdx.x * blockDim.x + threadIdx.x;
  long stride = (long)gridDim.x * blockDim.x;
  for (; i < n; i += stride) p[i] = 0.0f;
}

__global__ void k_zero_bf16(bf16_t* p, long n) {
  long i = (long)blockIdx.x * blockDim.x + threadIdx.x;
  long stride = (long)gridDim.x * blockDim.x;
  for (; i < n; i += stride) p[i] = (bf16_t)0.0f;
}

/* gather encoder embeddings -> bf16 [50][64][256] */
__global__ void k_gather_enc(const int* __restrict__ tok, const float* __restrict__ emb,
                             bf16_t* __restrict__ out) {
  int i = blockIdx.x * 256 + threadIdx.x;        /* 0 .. 50*64*256-1 */
  int c = i & 255;
  int sb = i >> 8;
  int t = tok[sb];
  out[i] = (bf16_t)emb[(size_t)t * EMB + c];
}

/* gather decoder embeddings directly into X columns [1536..1791] */
__global__ void k_gather_dec(const int* __restrict__ tok, const float* __restrict__ emb,
                             bf16_t* __restrict__ X) {
  int i = blockIdx.x * 256 + threadIdx.x;        /* 0 .. 49*64*256-1 */
  int c = i & 255;
  int row = i >> 8;                              /* t*64 + b */
  int t = tok[row];
  X[(size_t)row * KFC + 1536 + c] = (bf16_t)emb[(size_t)t * EMB + c];
}

/* ---------------- generic WMMA GEMM: C[MxN] = A[MxK] @ B[NxK]^T + bias ----------------
   one wave per 16x16 output tile; all M,N multiples of 16, K multiple of 32 */
__global__ void k_gemm(const bf16_t* __restrict__ A, int lda,
                       const bf16_t* __restrict__ B, int ldb,
                       const float* __restrict__ bias,
                       float* __restrict__ C, int ldc, int K, int act) {
  const int lane = threadIdx.x & 31;
  const int half = lane >> 4, r = lane & 15;
  const int n0 = blockIdx.x * 16, m0 = blockIdx.y * 16;
  v8f acc = {};
  const bf16_t* Ap = A + (size_t)(m0 + r) * lda + half * 16;
  const bf16_t* Bp = B + (size_t)(n0 + r) * ldb + half * 16;
  for (int k0 = 0; k0 < K; k0 += 32) {
    v8bf alo = *(const v8bf*)(Ap + k0);
    v8bf ahi = *(const v8bf*)(Ap + k0 + 8);
    v16bf av = __builtin_shufflevector(alo, ahi, 0,1,2,3,4,5,6,7,8,9,10,11,12,13,14,15);
    v16bf bv = *(const v16bf*)(Bp + k0);
    acc = __builtin_amdgcn_wmma_f32_16x16x32_bf16(false, av, false, bv,
                                                  (short)0, acc, false, false);
  }
  const int n = n0 + r;
  float bn = bias ? bias[n] : 0.0f;
#pragma unroll
  for (int v = 0; v < 8; ++v) {
    int m = m0 + half * 8 + v;
    float x = acc[v] + bn;
    if (act) x = tanhf(x);
    C[(size_t)m * ldc + n] = x;
  }
}

/* ---------------- big logits GEMM: X[3136x1792] @ Wfc[32000x1792]^T + bfc ----------------
   block = 8 waves, tile 64(M) x 256(N); wave = 4M x 2N 16x16 fragments.
   A double-buffered in LDS via async-to-LDS DMA (one barrier per K step);
   falls back to synchronous staging if the async builtins are unavailable. */
__global__ __launch_bounds__(256, 1)
void k_gemm_logits(const bf16_t* __restrict__ A, const bf16_t* __restrict__ B,
                   const float* __restrict__ bias, float* __restrict__ C) {
  __shared__ bf16_t As[2][64 * 32];
  const int tid  = threadIdx.x;
  const int wave = tid >> 5, lane = tid & 31;
  const int half = lane >> 4, r = lane & 15;
  const int M0 = blockIdx.y * 64;
  const int N0 = blockIdx.x * 256 + wave * 32;
  v8f acc[4][2] = {};
  const int arow  = tid >> 2;
  const int akoff = (tid & 3) * 8;
  const bf16_t* Ag = A + (size_t)(M0 + arow) * KFC + akoff;
  const bf16_t* B0 = B + (size_t)(N0 + r) * KFC + half * 16;
  const bf16_t* B1 = B + (size_t)(N0 + 16 + r) * KFC + half * 16;

#if HAVE_ASYNC_LDS
  /* prologue: DMA first A tile into buffer 0 (each thread moves 16B) */
  __builtin_amdgcn_global_load_async_to_lds_b128(
      (gas1_t)(Ag), (las3_t)(&As[0][arow * 32 + akoff]), 0, 0);
#endif

  int buf = 0;
  for (int k0 = 0; k0 < KFC; k0 += 32, buf ^= 1) {
#if HAVE_ASYNC_LDS
    wait_async0();
    __syncthreads();   /* tile k0 visible to all; all waves done reading buf^1 */
    if (k0 + 32 < KFC)
      __builtin_amdgcn_global_load_async_to_lds_b128(
          (gas1_t)(Ag + k0 + 32), (las3_t)(&As[buf ^ 1][arow * 32 + akoff]), 0, 0);
#else
    __syncthreads();
    *(uint4*)(&As[buf][arow * 32 + akoff]) = *(const uint4*)(Ag + k0);
    __syncthreads();
#endif
    v16bf bv0 = *(const v16bf*)(B0 + k0);
    v16bf bv1 = *(const v16bf*)(B1 + k0);
    if (k0 + 32 < KFC) {
      __builtin_prefetch(B0 + k0 + 32, 0, 3);
      __builtin_prefetch(B1 + k0 + 32, 0, 3);
    }
#pragma unroll
    for (int mt = 0; mt < 4; ++mt) {
      const bf16_t* Ap = &As[buf][(mt * 16 + r) * 32 + half * 16];
      v8bf alo = *(const v8bf*)(Ap);
      v8bf ahi = *(const v8bf*)(Ap + 8);
      v16bf av = __builtin_shufflevector(alo, ahi, 0,1,2,3,4,5,6,7,8,9,10,11,12,13,14,15);
      acc[mt][0] = __builtin_amdgcn_wmma_f32_16x16x32_bf16(false, av, false, bv0,
                                                           (short)0, acc[mt][0], false, false);
      acc[mt][1] = __builtin_amdgcn_wmma_f32_16x16x32_bf16(false, av, false, bv1,
                                                           (short)0, acc[mt][1], false, false);
    }
  }
  const int nl = N0 + r;
  const float b0 = bias[nl], b1 = bias[nl + 16];
#pragma unroll
  for (int mt = 0; mt < 4; ++mt)
#pragma unroll
    for (int v = 0; v < 8; ++v) {
      const int m = M0 + mt * 16 + half * 8 + v;
      C[(size_t)m * VOUT + nl]      = acc[mt][0][v] + b0;
      C[(size_t)m * VOUT + nl + 16] = acc[mt][1][v] + b1;
    }
}

/* ---------------- GRU cells ---------------- */

/* both encoder directions in one launch: idx = dir*32768 + b*512 + c */
__global__ void k_enc_cell(const float* __restrict__ giF, const float* __restrict__ ghF,
                           const float* __restrict__ giB, const float* __restrict__ ghB,
                           float* __restrict__ hF, bf16_t* __restrict__ hB,
                           bf16_t* __restrict__ encOut, bf16_t* __restrict__ hcat, int t) {
  int idx = blockIdx.x * 256 + threadIdx.x;
  int dir = idx >> 15;
  int rem = idx & 32767;
  int b = rem >> 9, c = rem & 511;
  const float* gi = dir ? giB : giF;
  const float* gh = dir ? ghB : ghF;
  float i0 = gi[b * 1536 + c],        h0 = gh[b * 1536 + c];
  float i1 = gi[b * 1536 + 512 + c],  h1 = gh[b * 1536 + 512 + c];
  float i2 = gi[b * 1536 + 1024 + c], h2 = gh[b * 1536 + 1024 + c];
  float h = hF[idx];
  float rg = sigmoidf_(i0 + h0);
  float z  = sigmoidf_(i1 + h1);
  float nn = tanhf(i2 + rg * h2);
  float hn = (1.0f - z) * nn + z * h;
  hF[idx] = hn;
  bf16_t hb = (bf16_t)hn;
  hB[idx] = hb;
  int s = dir ? (S_LEN - 1 - t) : t;
  encOut[((size_t)(s * BATCH + b)) * 1024 + dir * 512 + c] = hb;
  hcat[b * 1024 + dir * 512 + c] = hb;
}

__global__ void k_dec_cell(const float* __restrict__ gi, const float* __restrict__ gh,
                           float* __restrict__ hF, bf16_t* __restrict__ hB,
                           bf16_t* __restrict__ X, int t) {
  int idx = blockIdx.x * 256 + threadIdx.x;   /* 0..32767 */
  int b = idx >> 9, c = idx & 511;
  float i0 = gi[b * 1536 + c],        h0 = gh[b * 1536 + c];
  float i1 = gi[b * 1536 + 512 + c],  h1 = gh[b * 1536 + 512 + c];
  float i2 = gi[b * 1536 + 1024 + c], h2 = gh[b * 1536 + 1024 + c];
  float h = hF[idx];
  float rg = sigmoidf_(i0 + h0);
  float z  = sigmoidf_(i1 + h1);
  float nn = tanhf(i2 + rg * h2);
  float hn = (1.0f - z) * nn + z * h;
  hF[idx] = hn;
  bf16_t hb = (bf16_t)hn;
  hB[idx] = hb;
  X[((size_t)(t * BATCH + b)) * KFC + c] = hb;
}

/* ---------------- attention ---------------- */

/* one wave per (s,b): score = sum_c tanh(Eproj + hproj) * v  */
__global__ void k_attn_scores(const float* __restrict__ Eproj, const float* __restrict__ hproj,
                              const float* __restrict__ v, float* __restrict__ scores) {
  int w = blockIdx.x * 8 + (threadIdx.x >> 5);
  int lane = threadIdx.x & 31;
  int s = w >> 6, b = w & 63;
  const float* ep = Eproj + (size_t)(s * BATCH + b) * 512;
  const float* hp = hproj + b * 512;
  float p = 0.0f;
  for (int c = lane; c < 512; c += 32) p += tanhf(ep[c] + hp[c]) * v[c];
#pragma unroll
  for (int off = 16; off; off >>= 1) p += __shfl_down(p, off);
  if (lane == 0) scores[s * BATCH + b] = p;
}

/* per-b softmax over s + weighted sum over enc_out; fills Gin and X */
__global__ void k_attn_soft(const float* __restrict__ scores, const bf16_t* __restrict__ encOut,
                            bf16_t* __restrict__ X, bf16_t* __restrict__ Gin, int t) {
  __shared__ float sa[S_LEN];
  int b = blockIdx.x, tid = threadIdx.x;
  if (tid < S_LEN) sa[tid] = scores[tid * BATCH + b];
  __syncthreads();
  if (tid == 0) {
    float m = sa[0];
    for (int s = 1; s < S_LEN; ++s) m = fmaxf(m, sa[s]);
    float sum = 0.0f;
    for (int s = 0; s < S_LEN; ++s) { sa[s] = __expf(sa[s] - m); sum += sa[s]; }
    float inv = 1.0f / sum;
    for (int s = 0; s < S_LEN; ++s) sa[s] *= inv;
  }
  __syncthreads();
  size_t row = (size_t)(t * BATCH + b);
  for (int c = tid; c < 1024; c += 256) {
    float w = 0.0f;
    for (int s = 0; s < S_LEN; ++s)
      w += sa[s] * (float)encOut[((size_t)(s * BATCH + b)) * 1024 + c];
    bf16_t wb = (bf16_t)w;
    Gin[b * 1280 + 256 + c] = wb;          /* gru_in = [e, weighted] */
    X[row * KFC + 512 + c] = wb;           /* logits input = [h, weighted, e] */
  }
  Gin[b * 1280 + tid] = X[row * KFC + 1536 + tid];   /* copy e (tid<256) */
}

/* ---------------- host orchestration ---------------- */

extern "C" void kernel_launch(void* const* d_in, const int* in_sizes, int n_in,
                              void* d_out, int out_size, void* d_ws, size_t ws_size,
                              hipStream_t stream) {
  (void)in_sizes; (void)n_in; (void)out_size; (void)ws_size;
  const int*   source  = (const int*)  d_in[0];
  const int*   target  = (const int*)  d_in[1];
  const float* emb_enc = (const float*)d_in[2];
  const float* Wih_f   = (const float*)d_in[3];
  const float* Whh_f   = (const float*)d_in[4];
  const float* bih_f   = (const float*)d_in[5];
  const float* bhh_f   = (const float*)d_in[6];
  const float* Wih_b   = (const float*)d_in[7];
  const float* Whh_b   = (const float*)d_in[8];
  const float* bih_b   = (const float*)d_in[9];
  const float* bhh_b   = (const float*)d_in[10];
  const float* Wef     = (const float*)d_in[11];
  const float* bef     = (const float*)d_in[12];
  const float* Wa      = (const float*)d_in[13];
  const float* ba      = (const float*)d_in[14];
  const float* vvec    = (const float*)d_in[15];
  const float* emb_dec = (const float*)d_in[16];
  const float* Wih_d   = (const float*)d_in[17];
  const float* Whh_d   = (const float*)d_in[18];
  const float* bih_d   = (const float*)d_in[19];
  const float* bhh_d   = (const float*)d_in[20];
  const float* Wfc     = (const float*)d_in[21];
  const float* bfc     = (const float*)d_in[22];
  float* out = (float*)d_out;

  /* workspace carve-up (256B aligned) */
  char* p = (char*)d_ws;
  auto carve = [&](size_t bytes) -> char* {
    char* q = p; p += (bytes + 255) & ~(size_t)255; return q;
  };
  bf16_t* WfcB   = (bf16_t*)carve((size_t)VOUT * KFC * 2);
  bf16_t* WihdB  = (bf16_t*)carve((size_t)1536 * 1280 * 2);
  bf16_t* WhhdB  = (bf16_t*)carve((size_t)1536 * 512 * 2);
  bf16_t* WihfB  = (bf16_t*)carve((size_t)1536 * 256 * 2);
  bf16_t* WhhfB  = (bf16_t*)carve((size_t)1536 * 512 * 2);
  bf16_t* WihbB  = (bf16_t*)carve((size_t)1536 * 256 * 2);
  bf16_t* WhhbB  = (bf16_t*)carve((size_t)1536 * 512 * 2);
  bf16_t* WefB   = (bf16_t*)carve((size_t)512 * 1024 * 2);
  bf16_t* WaB    = (bf16_t*)carve((size_t)512 * 1536 * 2);
  bf16_t* embS   = (bf16_t*)carve((size_t)S_LEN * BATCH * EMB * 2);
  bf16_t* encOutB= (bf16_t*)carve((size_t)S_LEN * BATCH * 1024 * 2);
  float*  Eproj  = (float*) carve((size_t)S_LEN * BATCH * 512 * 4);
  float*  hF32   = (float*) carve((size_t)2 * BATCH * HENC * 4);
  bf16_t* hB16   = (bf16_t*)carve((size_t)2 * BATCH * HENC * 2);
  bf16_t* hcat   = (bf16_t*)carve((size_t)BATCH * 1024 * 2);
  float*  giF    = (float*) carve((size_t)BATCH * 1536 * 4);
  float*  ghF    = (float*) carve((size_t)BATCH * 1536 * 4);
  float*  giB    = (float*) carve((size_t)BATCH * 1536 * 4);
  float*  ghB    = (float*) carve((size_t)BATCH * 1536 * 4);
  float*  decHf  = (float*) carve((size_t)BATCH * 512 * 4);
  bf16_t* decHb  = (bf16_t*)carve((size_t)BATCH * 512 * 2);
  float*  hproj  = (float*) carve((size_t)BATCH * 512 * 4);
  float*  scores = (float*) carve((size_t)S_LEN * BATCH * 4);
  bf16_t* Gin    = (bf16_t*)carve((size_t)BATCH * 1280 * 2);
  bf16_t* X      = (bf16_t*)carve((size_t)49 * BATCH * KFC * 2);
  float*  giD    = (float*) carve((size_t)BATCH * 1536 * 4);
  float*  ghD    = (float*) carve((size_t)BATCH * 1536 * 4);

  auto cvt = [&](const float* s, bf16_t* d, long n) {
    k_cvt_bf16<<<dim3((unsigned)((n + 255) / 256)), dim3(256), 0, stream>>>(s, d, n);
  };
  auto gemm = [&](const bf16_t* A, int lda, const bf16_t* B, int ldb,
                  const float* bias, float* C, int ldc, int M, int N, int K, int act) {
    k_gemm<<<dim3(N / 16, M / 16), dim3(32), 0, stream>>>(A, lda, B, ldb, bias, C, ldc, K, act);
  };

  /* 1) weight conversion */
  cvt(Wfc,   WfcB,  (long)VOUT * KFC);
  cvt(Wih_d, WihdB, 1536L * 1280);
  cvt(Whh_d, WhhdB, 1536L * 512);
  cvt(Wih_f, WihfB, 1536L * 256);
  cvt(Whh_f, WhhfB, 1536L * 512);
  cvt(Wih_b, WihbB, 1536L * 256);
  cvt(Whh_b, WhhbB, 1536L * 512);
  cvt(Wef,   WefB,  512L * 1024);
  cvt(Wa,    WaB,   512L * 1536);

  /* 2) embedding gathers */
  k_gather_enc<<<dim3(S_LEN * BATCH * EMB / 256), dim3(256), 0, stream>>>(source, emb_enc, embS);
  k_gather_dec<<<dim3(49 * BATCH * EMB / 256), dim3(256), 0, stream>>>(target, emb_dec, X);

  /* 3) zero init */
  k_zero_f32 <<<dim3(256), dim3(256), 0, stream>>>(hF32, 2L * BATCH * HENC);
  k_zero_bf16<<<dim3(256), dim3(256), 0, stream>>>(hB16, 2L * BATCH * HENC);
  k_zero_f32 <<<dim3(8000), dim3(256), 0, stream>>>(out, (long)BATCH * VOUT);  /* t=0 row */

  /* 4) bidirectional encoder scan */
  for (int t = 0; t < S_LEN; ++t) {
    gemm(embS + (size_t)t * BATCH * EMB, EMB, WihfB, EMB, bih_f, giF, 1536, BATCH, 1536, EMB, 0);
    gemm(hB16, HENC, WhhfB, HENC, bhh_f, ghF, 1536, BATCH, 1536, HENC, 0);
    gemm(embS + (size_t)(S_LEN - 1 - t) * BATCH * EMB, EMB, WihbB, EMB, bih_b, giB, 1536, BATCH, 1536, EMB, 0);
    gemm(hB16 + (size_t)BATCH * HENC, HENC, WhhbB, HENC, bhh_b, ghB, 1536, BATCH, 1536, HENC, 0);
    k_enc_cell<<<dim3(2 * BATCH * HENC / 256), dim3(256), 0, stream>>>(
        giF, ghF, giB, ghB, hF32, hB16, encOutB, hcat, t);
  }

  /* 5) hidden0 = tanh([hf,hb] @ Wef^T + bef) */
  gemm(hcat, 1024, WefB, 1024, bef, decHf, 512, BATCH, 512, 1024, 1);
  cvt(decHf, decHb, (long)BATCH * 512);

  /* 6) attention precompute: Eproj = enc_out @ Wa[:,512:]^T + ba  */
  gemm(encOutB, 1024, WaB + 512, 1536, ba, Eproj, 512, S_LEN * BATCH, 512, 1024, 0);

  /* 7) decoder scan */
  for (int t = 0; t < 49; ++t) {
    gemm(decHb, 512, WaB, 1536, nullptr, hproj, 512, BATCH, 512, 512, 0);
    k_attn_scores<<<dim3(S_LEN * BATCH / 8), dim3(256), 0, stream>>>(Eproj, hproj, vvec, scores);
    k_attn_soft<<<dim3(BATCH), dim3(256), 0, stream>>>(scores, encOutB, X, Gin, t);
    gemm(Gin, 1280, WihdB, 1280, bih_d, giD, 1536, BATCH, 1536, 1280, 0);
    gemm(decHb, 512, WhhdB, 512, bhh_d, ghD, 1536, BATCH, 1536, 512, 0);
    k_dec_cell<<<dim3(BATCH * 512 / 256), dim3(256), 0, stream>>>(giD, ghD, decHf, decHb, X, t);
  }

  /* 8) batched output projection: out[1:] = X @ Wfc^T + bfc  (360 GFLOP WMMA GEMM) */
  k_gemm_logits<<<dim3(VOUT / 256, 49 * BATCH / 64), dim3(256), 0, stream>>>(
      X, WfcB, bfc, out + (size_t)BATCH * VOUT);
}
